// SelfAttentionHead_84851373900047
// MI455X (gfx1250) — compile-verified, matrix operand order checked by reading, not run on hardware
//
#include <hip/hip_runtime.h>
#include <cstdint>
#include <cstddef>

// ---------------------------------------------------------------------------
// SelfAttentionHead  B=4, S=4096, D=256  (fp32 in/out, bf16 WMMA internals)
// ---------------------------------------------------------------------------

#define D_MODEL 256
#define SEQ     4096
#define NBATCH  4
#define SCALE   0.0625f   // 1/sqrt(256)

typedef __attribute__((ext_vector_type(16))) __bf16 v16bf;
typedef __attribute__((ext_vector_type(8)))  float  v8f;
typedef unsigned short u16;
typedef unsigned int   u32;

union Frag16 {            // 16 bf16 elems = 8 VGPRs, matches WMMA A/B fragment
    v16bf v;
    u16   h[16];
    uint4 q[2];
};

__device__ __forceinline__ u16 f2bf(float f) {   // round-to-nearest-even
    u32 u = __float_as_uint(f);
    u32 r = u + 0x7FFFu + ((u >> 16) & 1u);
    return (u16)(r >> 16);
}
__device__ __forceinline__ u32 pack2(float a, float b) {
    return (u32)f2bf(a) | ((u32)f2bf(b) << 16);
}
__device__ __forceinline__ float rowmax16(float v) {  // reduce within 16-lane half
    v = fmaxf(v, __shfl_xor(v, 1, 32));
    v = fmaxf(v, __shfl_xor(v, 2, 32));
    v = fmaxf(v, __shfl_xor(v, 4, 32));
    v = fmaxf(v, __shfl_xor(v, 8, 32));
    return v;
}
__device__ __forceinline__ float rowsum16(float v) {
    v += __shfl_xor(v, 1, 32);
    v += __shfl_xor(v, 2, 32);
    v += __shfl_xor(v, 4, 32);
    v += __shfl_xor(v, 8, 32);
    return v;
}

// Async global->LDS copy of one 16-byte chunk per lane (ASYNCcnt-tracked DMA).
__device__ __forceinline__ void async_b128(u32 lds_addr, const void* gaddr) {
    asm volatile("global_load_async_to_lds_b128 %0, %1, off"
                 :: "v"(lds_addr), "v"(gaddr) : "memory");
}
__device__ __forceinline__ void wait_async0() {
    asm volatile("s_wait_asynccnt 0x0" ::: "memory");
}
__device__ __forceinline__ u32 lds_off(const void* p) {   // AS3 offset = low 32 bits
    return (u32)(size_t)p;
}

// --------------------------------------------------------------------------
// Wt[n*256+k] = bf16(W[k*256+n])  : weight transpose+convert (B-frag friendly)
// --------------------------------------------------------------------------
__global__ void wt_convert(const float* __restrict__ W, u16* __restrict__ Wt) {
    int n = blockIdx.x;
    int k = threadIdx.x;
    Wt[n * D_MODEL + k] = f2bf(W[k * D_MODEL + n]);
}

// --------------------------------------------------------------------------
// Projection: out = X @ W + b  for a 16-row tile.  One wave per block.
// MODE 0: write f32 (qp)    MODE 1: write bf16 row-major (kp)
// MODE 2: write bf16 transposed [B, D, S] (v^T)
// W fragments are double-buffered (parity-indexed, fully unrolled inner loop)
// so each WMMA only waits for the older outstanding load, not a full drain.
// --------------------------------------------------------------------------
template <int MODE>
__global__ __launch_bounds__(32)
void proj_kernel(const float* __restrict__ X, const u16* __restrict__ Wt,
                 const float* __restrict__ bias,
                 float* __restrict__ outF, u16* __restrict__ outB,
                 u16* __restrict__ outT)
{
    const int l  = threadIdx.x;
    const int ln = l & 15, lh = l >> 4;
    const int m0 = blockIdx.x * 16;

    // A fragments: 8 chunks of K=32 covering D=256, rows = lane%16
    Frag16 a[8];
    const float* xrow = X + (size_t)(m0 + ln) * D_MODEL;
#pragma unroll
    for (int ks = 0; ks < 8; ++ks) {
        int kb = ks * 32 + lh * 8;
#pragma unroll
        for (int e = 0; e < 8; ++e) {
            a[ks].h[e]     = f2bf(xrow[kb + e]);
            a[ks].h[8 + e] = f2bf(xrow[kb + 16 + e]);
        }
    }

    const u16* wbase = Wt + (size_t)ln * D_MODEL;   // + nt*16*256 per tile
    Frag16 bfr[2];
    // preload (nt=0, ks=0)
    bfr[0].q[0] = *(const uint4*)(wbase + lh * 8);
    bfr[0].q[1] = *(const uint4*)(wbase + lh * 8 + 16);

    for (int nt = 0; nt < 16; ++nt) {
        v8f c = {};
        const u16* wrow  = wbase + (size_t)nt * 16 * D_MODEL;
        const u16* wnext = wrow + 16 * D_MODEL;
#pragma unroll
        for (int ks = 0; ks < 8; ++ks) {
            // preload next fragment (depth-1 pipeline)
            if (ks < 7) {
                int kbn = (ks + 1) * 32 + lh * 8;
                bfr[(ks + 1) & 1].q[0] = *(const uint4*)(wrow + kbn);
                bfr[(ks + 1) & 1].q[1] = *(const uint4*)(wrow + kbn + 16);
            } else if (nt < 15) {
                bfr[0].q[0] = *(const uint4*)(wnext + lh * 8);
                bfr[0].q[1] = *(const uint4*)(wnext + lh * 8 + 16);
            }
            c = __builtin_amdgcn_wmma_f32_16x16x32_bf16(
                    false, a[ks].v, false, bfr[ks & 1].v, (short)0, c, false, false);
        }
        float bb  = bias[nt * 16 + ln];
        int   col = nt * 16 + ln;
        if (MODE == 0) {
#pragma unroll
            for (int r = 0; r < 8; ++r) {
                int row = m0 + r + 8 * lh;
                outF[(size_t)row * D_MODEL + col] = c[r] + bb;
            }
        } else if (MODE == 1) {
#pragma unroll
            for (int r = 0; r < 8; ++r) {
                int row = m0 + r + 8 * lh;
                outB[(size_t)row * D_MODEL + col] = f2bf(c[r] + bb);
            }
        } else {
            union { u16 pk[8]; uint4 q; } u;
#pragma unroll
            for (int r = 0; r < 8; ++r) u.pk[r] = f2bf(c[r] + bb);
            int b_ = m0 >> 12;                  // batch
            int s0 = (m0 & (SEQ - 1)) + 8 * lh; // 8 consecutive seq positions
            *(uint4*)(outT + (size_t)(b_ * D_MODEL + col) * SEQ + s0) = u.q;
        }
    }
}

// --------------------------------------------------------------------------
// Attention: 4-wave workgroup.  Each wave owns a 16-query-row tile; the block
// DMA-stages the shared 32-key K tile and V^T tile into double-buffered LDS
// with GLOBAL_LOAD_ASYNC_TO_LDS_B128 (ASYNCcnt), overlapping the next tile's
// transfer with this tile's WMMAs.  One barrier per j-step.
// Masked (j > i) scores are exactly 0 and STILL participate (faithful to the
// reference's multiplicative tril mask).
// --------------------------------------------------------------------------
__global__ __launch_bounds__(128)
void attn_kernel(const float* __restrict__ qpF, const u16* __restrict__ kB,
                 const u16* __restrict__ vT, float* __restrict__ out)
{
    __shared__ u16 qsh[4 * 16 * D_MODEL];     // 32 KiB : per-wave q tiles (bf16)
    __shared__ u16 ksh[2][32 * D_MODEL];      // 2 x 16 KiB : K tile, row-major
    __shared__ u16 vsh[2][D_MODEL * 32];      // 2 x 16 KiB : V^T tile [d][j]
    __shared__ u16 psh[4 * 16 * 32];          //  4 KiB : per-wave P tiles (bf16)

    const int tid = threadIdx.x;
    const int w   = tid >> 5;                 // wave id 0..3
    const int l   = tid & 31;
    const int ln  = l & 15, lh = l >> 4;
    const int g0  = (blockIdx.x * 4 + w) * 16;   // this wave's query-row base
    const int b_  = g0 >> 12;                    // batch (same for all 4 waves)
    const int i0  = g0 & (SEQ - 1);

    u16* qw = qsh + w * (16 * D_MODEL);
    u16* pw = psh + w * (16 * 32);

    // Issue the async DMA for one j-tile: K tile is a 16 KiB contiguous blob,
    // V^T tile is 256 rows x 64 B gathered from the [B,D,S] layout.
    auto stage_async = [&](int j0, int p) {
        const u16* kb0 = kB + (size_t)(b_ * SEQ + j0) * D_MODEL;
        u32 klds = lds_off(ksh[p]);
        u32 vlds = lds_off(vsh[p]);
#pragma unroll
        for (int c = 0; c < 8; ++c) {         // 1024 x 16B
            int idx = tid + 128 * c;
            async_b128(klds + idx * 16, kb0 + idx * 8);
        }
#pragma unroll
        for (int c = 0; c < 8; ++c) {         // 1024 x 16B
            int idx = tid + 128 * c;
            int d   = idx >> 2;               // 0..255
            int ch  = idx & 3;                // 4 x 8 bf16 per row
            async_b128(vlds + (d * 32 + ch * 8) * 2,
                       vT + (size_t)(b_ * D_MODEL + d) * SEQ + j0 + ch * 8);
        }
    };

    stage_async(0, 0);   // prologue: DMA first tile while we convert q below

    // ---- stage own q tile into LDS with f32->bf16 conversion (per wave) ----
    {
        const float4* src = (const float4*)(qpF + (size_t)g0 * D_MODEL);
        uint4* dst = (uint4*)qw;
#pragma unroll
        for (int t = 0; t < 16; ++t) {
            int idx = l + 32 * t;            // 0..511, 8 bf16 per idx
            float4 x0 = src[idx * 2];
            float4 x1 = src[idx * 2 + 1];
            uint4 o;
            o.x = pack2(x0.x, x0.y); o.y = pack2(x0.z, x0.w);
            o.z = pack2(x1.x, x1.y); o.w = pack2(x1.z, x1.w);
            dst[idx] = o;
        }
    }

    v8f acc[16];
#pragma unroll
    for (int dt = 0; dt < 16; ++dt) acc[dt] = (v8f){};
    float mrun[8], srun[8];
#pragma unroll
    for (int r = 0; r < 8; ++r) { mrun[r] = -__builtin_inff(); srun[r] = 0.f; }

    for (int jt = 0; jt < SEQ / 32; ++jt) {
        const int j0 = jt * 32;
        const int p  = jt & 1;

        wait_async0();       // my DMA writes into buffer p are in LDS
        __syncthreads();     // everyone's DMA writes visible; buf p^1 free

        if (jt + 1 < SEQ / 32)
            stage_async(j0 + 32, p ^ 1);   // DMA next tile during compute

        const u16* kp_ = ksh[p];
        const u16* vp_ = vsh[p];

        // ---- scores S = q . k^T  (two 16x16 C-frags over K=256) ----
        v8f sc0 = {}, sc1 = {};
#pragma unroll
        for (int ks = 0; ks < 8; ++ks) {
            int kb = ks * 32 + lh * 8;
            Frag16 a, b0, b1;
            a.q[0]  = *(const uint4*)(qw + ln * D_MODEL + kb);
            a.q[1]  = *(const uint4*)(qw + ln * D_MODEL + kb + 16);
            b0.q[0] = *(const uint4*)(kp_ + ln * D_MODEL + kb);
            b0.q[1] = *(const uint4*)(kp_ + ln * D_MODEL + kb + 16);
            b1.q[0] = *(const uint4*)(kp_ + (16 + ln) * D_MODEL + kb);
            b1.q[1] = *(const uint4*)(kp_ + (16 + ln) * D_MODEL + kb + 16);
            sc0 = __builtin_amdgcn_wmma_f32_16x16x32_bf16(
                      false, a.v, false, b0.v, (short)0, sc0, false, false);
            sc1 = __builtin_amdgcn_wmma_f32_16x16x32_bf16(
                      false, a.v, false, b1.v, (short)0, sc1, false, false);
        }

        // ---- mask (multiplicative!), online softmax update ----
        float corr[8], p0[8], p1[8];
#pragma unroll
        for (int r = 0; r < 8; ++r) {
            int   row = i0 + r + 8 * lh;
            float s0v = ((j0 + ln)      <= row) ? sc0[r] * SCALE : 0.0f;
            float s1v = ((j0 + 16 + ln) <= row) ? sc1[r] * SCALE : 0.0f;
            float t   = rowmax16(fmaxf(s0v, s1v));
            float mn  = fmaxf(mrun[r], t);
            float co  = __expf(mrun[r] - mn);
            float e0  = __expf(s0v - mn);
            float e1  = __expf(s1v - mn);
            srun[r]   = srun[r] * co + rowsum16(e0 + e1);
            mrun[r]   = mn;
            corr[r]   = co;
            p0[r] = e0; p1[r] = e1;
        }
#pragma unroll
        for (int dt = 0; dt < 16; ++dt)
#pragma unroll
            for (int r = 0; r < 8; ++r) acc[dt][r] *= corr[r];

        // ---- C-layout -> A-layout transpose of P through per-wave LDS ----
        // (same-wave DS ops are in-order: no block barrier needed)
#pragma unroll
        for (int r = 0; r < 8; ++r) {
            pw[(r + 8 * lh) * 32 + ln]      = f2bf(p0[r]);
            pw[(r + 8 * lh) * 32 + 16 + ln] = f2bf(p1[r]);
        }
        Frag16 pa;
        pa.q[0] = *(const uint4*)(pw + ln * 32 + lh * 8);
        pa.q[1] = *(const uint4*)(pw + ln * 32 + 16 + lh * 8);

        // ---- acc += P (16x32) x V (32x16) per 16-wide d-tile ----
#pragma unroll
        for (int dt = 0; dt < 16; ++dt) {
            Frag16 vb;
            const u16* vp = vp_ + (dt * 16 + ln) * 32 + lh * 8;
            vb.q[0] = *(const uint4*)vp;
            vb.q[1] = *(const uint4*)(vp + 16);
            acc[dt] = __builtin_amdgcn_wmma_f32_16x16x32_bf16(
                          false, pa.v, false, vb.v, (short)0, acc[dt], false, false);
        }
    }

    // ---- epilogue: out = qp + acc / Z ----
#pragma unroll
    for (int dt = 0; dt < 16; ++dt) {
        int col = dt * 16 + ln;
#pragma unroll
        for (int r = 0; r < 8; ++r) {
            int row = g0 + r + 8 * lh;
            out[(size_t)row * D_MODEL + col] =
                qpF[(size_t)row * D_MODEL + col] + acc[dt][r] / srun[r];
        }
    }
}

// --------------------------------------------------------------------------
extern "C" void kernel_launch(void* const* d_in, const int* in_sizes, int n_in,
                              void* d_out, int out_size, void* d_ws, size_t ws_size,
                              hipStream_t stream) {
    (void)in_sizes; (void)n_in; (void)out_size; (void)ws_size;
    const float* v  = (const float*)d_in[0];
    const float* k  = (const float*)d_in[1];
    const float* q  = (const float*)d_in[2];
    const float* Wq = (const float*)d_in[3];
    const float* bq = (const float*)d_in[4];
    const float* Wk = (const float*)d_in[5];
    const float* bk = (const float*)d_in[6];
    const float* Wv = (const float*)d_in[7];
    const float* bv = (const float*)d_in[8];
    float* out = (float*)d_out;

    char* ws   = (char*)d_ws;
    float* qpF = (float*)(ws);                         // 16 MiB  fp32 qp
    u16*   kB  = (u16*)(ws + (16u << 20));             //  8 MiB  bf16 kp
    u16*   vT  = (u16*)(ws + (24u << 20));             //  8 MiB  bf16 v^T [B,D,S]
    u16*   WtQ = (u16*)(ws + (32u << 20));             //  3 x 128 KiB bf16 W^T
    u16*   WtK = WtQ + D_MODEL * D_MODEL;
    u16*   WtV = WtK + D_MODEL * D_MODEL;

    wt_convert<<<D_MODEL, D_MODEL, 0, stream>>>(Wq, WtQ);
    wt_convert<<<D_MODEL, D_MODEL, 0, stream>>>(Wk, WtK);
    wt_convert<<<D_MODEL, D_MODEL, 0, stream>>>(Wv, WtV);

    const int ntiles = NBATCH * SEQ / 16;              // 1024
    proj_kernel<0><<<ntiles, 32, 0, stream>>>(q, WtQ, bq, qpF, nullptr, nullptr);
    proj_kernel<1><<<ntiles, 32, 0, stream>>>(k, WtK, bk, nullptr, kB,  nullptr);
    proj_kernel<2><<<ntiles, 32, 0, stream>>>(v, WtV, bv, nullptr, nullptr, vT);

    attn_kernel<<<ntiles / 4, 128, 0, stream>>>(qpF, kB, vT, out);
}
